// Attention_64561948394150
// MI455X (gfx1250) — matrix operand order repaired and checked by reading, not compiled, not run on hardware
//
#include <hip/hip_runtime.h>
#include <hip/hip_bf16.h>

typedef __bf16 bf16_t;
typedef __attribute__((ext_vector_type(16))) __bf16 v16bf;
typedef __attribute__((ext_vector_type(8)))  __bf16 v8bf;
typedef __attribute__((ext_vector_type(8)))  float   v8f;

#define L_SEQ  2048
#define BATCH  2
#define EMB    256
#define HEADS  8
#define HDIM   32
#define ROWS   (L_SEQ * BATCH)          // 4096
// SCALE * log2(e): softmax done in exp2 domain (v_exp_f32 is natively base-2)
#define SCALE_Q (0.17677669529663687f * 1.4426950408889634f)
#define WPK_ELEMS (EMB * EMB)           // 65536 bf16 per packed weight

// ---- helpers --------------------------------------------------------------

__device__ __forceinline__ bf16_t f2bf(float f) { return static_cast<bf16_t>(f); }

// native v_exp_f32 (base-2 exponent)
__device__ __forceinline__ float fast_exp2(float x) {
  return __builtin_amdgcn_exp2f(x);
}

__device__ __forceinline__ v16bf cat16(v8bf lo, v8bf hi) {
  return __builtin_shufflevector(lo, hi, 0,1,2,3,4,5,6,7,8,9,10,11,12,13,14,15);
}

__device__ __forceinline__ v8f zero8() {
  return (v8f){0.f,0.f,0.f,0.f,0.f,0.f,0.f,0.f};
}

// A-fragment (16x32 bf16, MxK) from a row-major bf16 matrix, ld multiple of 8.
// lane m = row; elements: k = h*16+e
__device__ __forceinline__ v16bf loadA_bf16(const bf16_t* __restrict__ base,
                                            int ld, int lane) {
  const int m = lane & 15, h = lane >> 4;
  const bf16_t* p = base + (size_t)m * ld + h * 16;
  v8bf lo = *(const v8bf*)p;
  v8bf hi = *(const v8bf*)(p + 8);
  return cat16(lo, hi);
}

// B-fragment from pre-packed weights: one contiguous 32B block per lane.
__device__ __forceinline__ v16bf loadB_packed(const bf16_t* __restrict__ wpk,
                                              int tG, int kidx, int lane) {
  const bf16_t* p = wpk + ((size_t)((tG * 8 + kidx) * 32 + lane)) * 16;
  return cat16(*(const v8bf*)p, *(const v8bf*)(p + 8));
}

// Async-copy one 2KB chunk (32 rows x 32 bf16) global -> LDS.
// 4 x GLOBAL_LOAD_ASYNC_TO_LDS_B128 (GV mode), 32 lanes x 16B each.
// Tracked by ASYNCcnt; consumer must s_wait_asynccnt.
__device__ __forceinline__ void async_copy_2k(const bf16_t* gsrc,
                                              bf16_t* lds_dst, int lane) {
  uint32_t lofs = (uint32_t)(uintptr_t)lds_dst;     // addr[31:0] = LDS offset
  uint64_t g    = (uint64_t)(uintptr_t)gsrc;
#pragma unroll
  for (int i = 0; i < 4; ++i) {
    uint32_t lo = lofs + (uint32_t)(i * 512 + lane * 16);
    uint64_t go = g    + (uint64_t)(i * 512 + lane * 16);
    asm volatile("global_load_async_to_lds_b128 %0, %1, off"
                 :: "v"(lo), "v"(go) : "memory");
  }
}

// ---- kernel 0: pack W (f32 row-major [256,256]) into bf16 B-fragment order
// dst[((t*8+kc)*32 + lane)*16 + e] = W[(kc*32 + h2*16 + e)*256 + t*16 + n]

__global__ __launch_bounds__(256)
void pack_w_kernel(const float* __restrict__ W, bf16_t* __restrict__ dst) {
  const int tid  = blockIdx.x * 256 + threadIdx.x;       // 0..4095
  const int lane = tid & 31;
  const int frag = tid >> 5;                             // 0..127 = t*8 + kc
  const int t = frag >> 3, kc = frag & 7;
  const int n = lane & 15, h2 = lane >> 4;
  const float* src = W + (size_t)(kc * 32 + h2 * 16) * EMB + t * 16 + n;
  bf16_t* out = dst + (size_t)tid * 16;
  v8bf a, b;
#pragma unroll
  for (int e = 0; e < 8; ++e) {
    a[e] = f2bf(src[(size_t)e * EMB]);
    b[e] = f2bf(src[(size_t)(e + 8) * EMB]);
  }
  *(v8bf*)out = a;
  *(v8bf*)(out + 8) = b;
}

// ---- kernel 1: X(f32)[4096,256] @ Wpk + b -> head-layout bf16 -------------
// TRANSPOSE_OUT=false: out[b][h][l][d] (Q, K);  true: out[b][h][d][l] (V^T)

template <bool TRANSPOSE_OUT>
__global__ __launch_bounds__(128)
void proj_f32_kernel(const float* __restrict__ X, const bf16_t* __restrict__ Wpk,
                     const float* __restrict__ bias, bf16_t* __restrict__ out,
                     float scale) {
  const int lane  = threadIdx.x & 31;
  const int wave  = threadIdx.x >> 5;
  const int n     = lane & 15;
  const int h2    = lane >> 4;
  const int rBase = blockIdx.x * 64 + wave * 16;
  const int tBase = blockIdx.y * 4;              // 16-col tile index base

  v8f acc[4];
#pragma unroll
  for (int t = 0; t < 4; ++t) acc[t] = zero8();

  for (int kidx = 0; kidx < 8; ++kidx) {
    // A fragment: f32 -> bf16 (hardware cvt), loads vectorize to b128
    const float* ap = X + (size_t)(rBase + n) * EMB + kidx * 32 + h2 * 16;
    v16bf A;
#pragma unroll
    for (int e = 0; e < 8; ++e) { A[e] = f2bf(ap[e]); A[e + 8] = f2bf(ap[8 + e]); }
#pragma unroll
    for (int t = 0; t < 4; ++t) {
      v16bf Bv = loadB_packed(Wpk, tBase + t, kidx, lane);
      acc[t] = __builtin_amdgcn_wmma_f32_16x16x32_bf16(
          false, A, false, Bv, (short)0, acc[t], false, false);
    }
  }
#pragma unroll
  for (int t = 0; t < 4; ++t) {
    const int col = (tBase + t) * 16 + n;
    const int hh = col >> 5, dd = col & 31;
    const float bcol = bias[col];
#pragma unroll
    for (int r = 0; r < 8; ++r) {
      const int row = rBase + r + 8 * h2;        // row = l*BATCH + b
      const int l = row >> 1, b = row & 1;
      const bf16_t v = f2bf((acc[t][r] + bcol) * scale);
      if (TRANSPOSE_OUT)
        out[((size_t)(b * HEADS + hh) * HDIM + dd) * L_SEQ + l] = v;   // [bh][d][l]
      else
        out[((size_t)(b * HEADS + hh) * L_SEQ + l) * HDIM + dd] = v;   // [bh][l][d]
    }
  }
}

// ---- kernel 2: flash attention, S^T = K Q^T, O^T = V^T P^T ----------------
// Barrier-free. K chunks double-buffered into wave-private LDS with
// GLOBAL_LOAD_ASYNC_TO_LDS_B128 (ASYNCcnt); V^T straight from global;
// P^T staged per-wave in LDS directly in B-fragment order.

__global__ __launch_bounds__(128)
void attn_kernel(const bf16_t* __restrict__ Qh, const bf16_t* __restrict__ Kh,
                 const bf16_t* __restrict__ VhT, bf16_t* __restrict__ X2) {
  __shared__ __attribute__((aligned(16))) bf16_t Ps[4 * 32 * 16];        // 4KB
  __shared__ __attribute__((aligned(16))) bf16_t Kst[4][2][32 * HDIM];   // 16KB

  const int tid  = threadIdx.x;
  const int lane = tid & 31, wave = tid >> 5;
  const int n = lane & 15, h2 = lane >> 4;
  const int bh = blockIdx.y;                 // b*HEADS + h
  const int b = bh / HEADS, hh = bh % HEADS;
  const int qBase = blockIdx.x * 64 + wave * 16;

  const bf16_t* Qb  = Qh  + (size_t)bh * L_SEQ * HDIM;
  const bf16_t* Kb  = Kh  + (size_t)bh * L_SEQ * HDIM;
  const bf16_t* VbT = VhT + (size_t)bh * HDIM * L_SEQ;   // [d][l]

  // Q^T as B operand (lane n = query column, elems = d), constant over loop
  const bf16_t* qp = Qb + (size_t)(qBase + n) * HDIM + h2 * 16;
  v16bf Bq = cat16(*(const v8bf*)qp, *(const v8bf*)(qp + 8));

  v8f Ot0 = zero8(), Ot1 = zero8();
  float m_run = -3.0e38f, l_run = 0.f;
  bf16_t* PsW = &Ps[wave * 32 * 16];
  bf16_t* Kbuf0 = &Kst[wave][0][0];
  bf16_t* Kbuf1 = &Kst[wave][1][0];

  // preload chunk 0 asynchronously
  async_copy_2k(Kb, Kbuf0, lane);

  for (int jc = 0; jc < L_SEQ; jc += 32) {
    const int cur = (jc >> 5) & 1;
    bf16_t* Kcur = cur ? Kbuf1 : Kbuf0;
    if (jc + 32 < L_SEQ) {
      // issue next chunk, then wait only for the current one (<=4 in flight)
      async_copy_2k(Kb + (size_t)(jc + 32) * HDIM, cur ? Kbuf0 : Kbuf1, lane);
      __builtin_prefetch(VbT + jc + 32 + lane, 0, 1);
      asm volatile("s_wait_asynccnt 0x4" ::: "memory");
    } else {
      asm volatile("s_wait_asynccnt 0x0" ::: "memory");
    }

    // S^T tiles: [j, q] = K-tile (A, from LDS) x Q^T (B)
    v16bf Ak0 = loadA_bf16(Kcur,              HDIM, lane);
    v16bf Ak1 = loadA_bf16(Kcur + 16 * HDIM,  HDIM, lane);
    v8f S0 = __builtin_amdgcn_wmma_f32_16x16x32_bf16(false, Ak0, false, Bq,
                                                     (short)0, zero8(), false, false);
    v8f S1 = __builtin_amdgcn_wmma_f32_16x16x32_bf16(false, Ak1, false, Bq,
                                                     (short)0, zero8(), false, false);
    // online softmax (exp2 domain) per query column (lane pair n, n+16)
    float tm = S0[0];
#pragma unroll
    for (int r = 1; r < 8; ++r) tm = fmaxf(tm, S0[r]);
#pragma unroll
    for (int r = 0; r < 8; ++r) tm = fmaxf(tm, S1[r]);
    tm = fmaxf(tm, __shfl_xor(tm, 16, 32));
    const float m_new = fmaxf(m_run, tm);
    const float corr  = fast_exp2(m_run - m_new);
    float rs = 0.f;
#pragma unroll
    for (int r = 0; r < 8; ++r) {
      S0[r] = fast_exp2(S0[r] - m_new); rs += S0[r];
      S1[r] = fast_exp2(S1[r] - m_new); rs += S1[r];
    }
    rs += __shfl_xor(rs, 16, 32);
    l_run = l_run * corr + rs;
    m_run = m_new;
#pragma unroll
    for (int r = 0; r < 8; ++r) { Ot0[r] *= corr; Ot1[r] *= corr; }

    // stage P^T directly in B-fragment order (two 16B LDS stores per lane)
    v8bf p0, p1;
#pragma unroll
    for (int r = 0; r < 8; ++r) { p0[r] = f2bf(S0[r]); p1[r] = f2bf(S1[r]); }
    *(v8bf*)(PsW + n * 16 + 8 * h2)        = p0;
    *(v8bf*)(PsW + (n + 16) * 16 + 8 * h2) = p1;
    asm volatile("s_wait_dscnt 0x0" ::: "memory");   // same-wave LDS ordering
    const bf16_t* pr = PsW + lane * 16;
    v16bf Bp = cat16(*(const v8bf*)pr, *(const v8bf*)(pr + 8));

    // O^T += V^T-tile (A, straight from global [d][l]) x P^T (B)
    v16bf Av0 = loadA_bf16(VbT + jc,                      L_SEQ, lane);
    v16bf Av1 = loadA_bf16(VbT + (size_t)16 * L_SEQ + jc, L_SEQ, lane);
    Ot0 = __builtin_amdgcn_wmma_f32_16x16x32_bf16(false, Av0, false, Bp,
                                                  (short)0, Ot0, false, false);
    Ot1 = __builtin_amdgcn_wmma_f32_16x16x32_bf16(false, Av1, false, Bp,
                                                  (short)0, Ot1, false, false);
  }

  // finalize: O[q,d] = O^T / l ; contiguous 16B bf16 stores
  const float inv = 1.0f / l_run;
  const size_t orow = ((size_t)(qBase + n) * BATCH + b) * EMB + hh * HDIM;
  v8bf o0, o1;
#pragma unroll
  for (int r = 0; r < 8; ++r) { o0[r] = f2bf(Ot0[r] * inv); o1[r] = f2bf(Ot1[r] * inv); }
  *(v8bf*)(X2 + orow + h2 * 8)      = o0;   // d = 0..15 half
  *(v8bf*)(X2 + orow + 16 + h2 * 8) = o1;   // d = 16..31 half
}

// ---- kernel 3: X2(bf16)[4096,256] @ Wp_packed + bp -> f32 out [L,B,E] -----

__global__ __launch_bounds__(128)
void out_proj_kernel(const bf16_t* __restrict__ X, const bf16_t* __restrict__ Wpk,
                     const float* __restrict__ bias, float* __restrict__ out) {
  const int lane  = threadIdx.x & 31;
  const int wave  = threadIdx.x >> 5;
  const int n     = lane & 15;
  const int h2    = lane >> 4;
  const int rBase = blockIdx.x * 64 + wave * 16;
  const int tBase = blockIdx.y * 4;

  v8f acc[4];
#pragma unroll
  for (int t = 0; t < 4; ++t) acc[t] = zero8();

  for (int kidx = 0; kidx < 8; ++kidx) {
    v16bf A = loadA_bf16(X + (size_t)rBase * EMB + kidx * 32, EMB, lane);
#pragma unroll
    for (int t = 0; t < 4; ++t) {
      v16bf Bv = loadB_packed(Wpk, tBase + t, kidx, lane);
      acc[t] = __builtin_amdgcn_wmma_f32_16x16x32_bf16(
          false, A, false, Bv, (short)0, acc[t], false, false);
    }
  }
#pragma unroll
  for (int t = 0; t < 4; ++t) {
    const int col = (tBase + t) * 16 + n;
    const float bcol = bias[col];
#pragma unroll
    for (int r = 0; r < 8; ++r) {
      const int row = rBase + r + 8 * h2;
      out[(size_t)row * EMB + col] = acc[t][r] + bcol;
    }
  }
}

// ---- launcher -------------------------------------------------------------

extern "C" void kernel_launch(void* const* d_in, const int* in_sizes, int n_in,
                              void* d_out, int out_size, void* d_ws, size_t ws_size,
                              hipStream_t stream) {
  const float* query = (const float*)d_in[0];
  const float* key_  = (const float*)d_in[1];
  const float* value = (const float*)d_in[2];
  const float* Wq = (const float*)d_in[3];
  const float* bq = (const float*)d_in[4];
  const float* Wk = (const float*)d_in[5];
  const float* bk = (const float*)d_in[6];
  const float* Wv = (const float*)d_in[7];
  const float* bv = (const float*)d_in[8];
  const float* Wp = (const float*)d_in[9];
  const float* bp = (const float*)d_in[10];

  const size_t NE = (size_t)BATCH * HEADS * L_SEQ * HDIM;  // 1,048,576 elems
  bf16_t* Wqp = (bf16_t*)d_ws;                 // 4 x 65536 bf16 = 512 KB
  bf16_t* Wkp = Wqp + WPK_ELEMS;
  bf16_t* Wvp = Wkp + WPK_ELEMS;
  bf16_t* Wpp = Wvp + WPK_ELEMS;
  bf16_t* Qh  = Wpp + WPK_ELEMS;               // 4 x 2 MB
  bf16_t* Kh  = Qh + NE;
  bf16_t* VhT = Kh + NE;                       // [b][h][d][l]
  bf16_t* X2  = VhT + NE;

  dim3 gPack(16);
  dim3 bPack(256);
  dim3 gProj(ROWS / 64, EMB / 64);          // 64 x 4
  dim3 gAttn(L_SEQ / 64, BATCH * HEADS);    // 32 x 16
  dim3 blk(128);

  pack_w_kernel<<<gPack, bPack, 0, stream>>>(Wq, Wqp);
  pack_w_kernel<<<gPack, bPack, 0, stream>>>(Wk, Wkp);
  pack_w_kernel<<<gPack, bPack, 0, stream>>>(Wv, Wvp);
  pack_w_kernel<<<gPack, bPack, 0, stream>>>(Wp, Wpp);

  proj_f32_kernel<false><<<gProj, blk, 0, stream>>>(query, Wqp, bq, Qh, SCALE_Q);
  proj_f32_kernel<false><<<gProj, blk, 0, stream>>>(key_,  Wkp, bk, Kh, 1.0f);
  proj_f32_kernel<true ><<<gProj, blk, 0, stream>>>(value, Wvp, bv, VhT, 1.0f);

  attn_kernel<<<gAttn, blk, 0, stream>>>(Qh, Kh, VhT, X2);
  out_proj_kernel<<<gProj, blk, 0, stream>>>(X2, Wpp, bp, (float*)d_out);
}
